// MultiheadDiffAttn_74869869904260
// MI455X (gfx1250) — compile-verified, matrix operand order checked by reading, not run on hardware
//
#include <hip/hip_runtime.h>

typedef __attribute__((ext_vector_type(16))) __bf16 bf16x16;
typedef __attribute__((ext_vector_type(8)))  float  f32x8;
typedef __attribute__((ext_vector_type(4)))  int    i32x4;
typedef __attribute__((ext_vector_type(8)))  int    i32x8;
typedef __attribute__((ext_vector_type(4)))  unsigned int u32x4;

#define E_DIM 2048
#define T_DIM 2048
#define H_NUM 16
#define LAMBDA_INIT_F 0.7836057665f

#define AS1 __attribute__((address_space(1)))
#define AS3 __attribute__((address_space(3)))

#if __has_builtin(__builtin_amdgcn_global_load_async_to_lds_b128) && \
    __has_builtin(__builtin_amdgcn_s_wait_asynccnt)
#define USE_ASYNC_LDS 1
#else
#define USE_ASYNC_LDS 0
#endif

#if __has_builtin(__builtin_amdgcn_tensor_load_to_lds) && \
    __has_builtin(__builtin_amdgcn_s_wait_tensorcnt)
#define USE_TDM 1
#else
#define USE_TDM 0
#endif

static __device__ __forceinline__ unsigned short f2bf(float f) {
    unsigned int u = __float_as_uint(f);
    unsigned int r = u + 0x7FFFu + ((u >> 16) & 1u);  // round-to-nearest-even
    return (unsigned short)(r >> 16);
}

// 16-element bf16 WMMA fragment: lane holds k = {kb..kb+7, kb+16..kb+23}
static __device__ __forceinline__ bf16x16 load_frag(const unsigned short* p) {
    union { uint4 u[2]; bf16x16 v; } r;
    r.u[0] = *reinterpret_cast<const uint4*>(p);
    r.u[1] = *reinterpret_cast<const uint4*>(p + 16);
    return r.v;
}

static __device__ __forceinline__ f32x8 wmma_bf16(bf16x16 a, bf16x16 b, f32x8 c) {
    return __builtin_amdgcn_wmma_f32_16x16x32_bf16(false, a, false, b, (short)0, c,
                                                   false, false);
}

// async global->LDS 16B copy (per-lane), sync fallback
static __device__ __forceinline__ void async_cp16(const unsigned short* g,
                                                  unsigned short* l) {
#if USE_ASYNC_LDS
    __builtin_amdgcn_global_load_async_to_lds_b128((AS1 i32x4*)g, (AS3 i32x4*)l,
                                                   0, 0);
#else
    *(uint4*)l = *(const uint4*)g;
#endif
}
static __device__ __forceinline__ void async_wait_all() {
#if USE_ASYNC_LDS
    __builtin_amdgcn_s_wait_asynccnt(0);
#endif
}

// TDM: load 32x64 bf16 tile (global row stride 2048 elems) into LDS laid out
// as [32][80] u16 (pad 8 DWORDs after every 32 DWORDs). One wave issues this.
static __device__ __forceinline__ void tdm_load_k(const unsigned short* gsrc,
                                                  unsigned ldsOff) {
#if USE_TDM
    unsigned long long ga = (unsigned long long)(size_t)gsrc;
    u32x4 g0;
    g0[0] = 1u;                                   // count=1, user descriptor
    g0[1] = ldsOff;                               // lds_addr [63:32]
    g0[2] = (unsigned)(ga & 0xFFFFFFFFu);         // global_addr low
    g0[3] = (unsigned)((ga >> 32) & 0x01FFFFFFu)  // global_addr[56:32]
            | 0x80000000u;                        // type=2 ("image")
    i32x8 g1;
    g1[0] = (1 << 16) | (1 << 20) | (4 << 22) | (7 << 25);
    //        data=2B   pad_en      intv=32dw    amount=8dw
    g1[1] = (int)(64u << 16);                     // tensor_dim0 = 64
    g1[2] = (int)(32u << 16);                     // tensor_dim1 = 32
    g1[3] = (int)(64u << 16);                     // tile_dim0 = 64
    g1[4] = 32;                                   // tile_dim1 = 32, tile_dim2 = 0
    g1[5] = 2048;                                 // tensor_dim0_stride = 2048
    g1[6] = 0;
    g1[7] = 0;
    i32x4 g2; g2[0] = 1; g2[1] = 1; g2[2] = 0; g2[3] = 0;   // dim2=dim3=1
    i32x4 g3; g3[0] = 0; g3[1] = (int)(1u << 16); g3[2] = 0; g3[3] = 0; // dim4=1
    i32x8 g4;
    for (int i = 0; i < 8; i++) g4[i] = 0;        // aux group (zero-filled)
    __builtin_amdgcn_tensor_load_to_lds(g0, g1, g2, g3, g4, 0);
#else
    (void)gsrc; (void)ldsOff;
#endif
}
static __device__ __forceinline__ void tdm_wait_all() {
#if USE_TDM
    __builtin_amdgcn_s_wait_tensorcnt(0);
#endif
}

// xor-shuffle across the 16-lane half via ds_swizzle
template <int MASK>
static __device__ __forceinline__ float xor16f(float v) {
#if __has_builtin(__builtin_amdgcn_ds_swizzle)
    return __int_as_float(
        __builtin_amdgcn_ds_swizzle(__float_as_int(v), (MASK << 10) | 0x1F));
#else
    return __shfl_xor(v, MASK);
#endif
}
static __device__ __forceinline__ float redmax16(float v) {
    v = fmaxf(v, xor16f<1>(v)); v = fmaxf(v, xor16f<2>(v));
    v = fmaxf(v, xor16f<4>(v)); v = fmaxf(v, xor16f<8>(v));
    return v;
}
static __device__ __forceinline__ float redsum16(float v) {
    v += xor16f<1>(v); v += xor16f<2>(v);
    v += xor16f<4>(v); v += xor16f<8>(v);
    return v;
}

// ---------------------------------------------------------------------------
__global__ void cast_f32_bf16(const float* __restrict__ in,
                              unsigned short* __restrict__ out, int n) {
    int i = blockIdx.x * blockDim.x + threadIdx.x;
    if (i < n) out[i] = f2bf(in[i]);
}

// ---------------------------------------------------------------------------
// C[M,2048] = A[M,2048] * W[2048,2048]^T   (bf16 in, fp32 accum)
// ---------------------------------------------------------------------------
template <int F32OUT>
__global__ __launch_bounds__(128) void gemm_bf16(
    const unsigned short* __restrict__ A, const unsigned short* __restrict__ W,
    void* __restrict__ Cout)
{
    const int GK = E_DIM, GN = E_DIM;
    const int lane = threadIdx.x & 31;
    const int wave = threadIdx.x >> 5;
    const int wm = wave >> 1, wn = wave & 1;
    const int mBase = blockIdx.y * 128 + wm * 64;
    const int nBase = blockIdx.x * 128 + wn * 64;
    const int l15 = lane & 15;
    const int kb  = (lane >> 4) << 3;

    f32x8 acc[4][4];
    for (int i = 0; i < 4; i++)
        for (int j = 0; j < 4; j++)
            for (int r = 0; r < 8; r++) acc[i][j][r] = 0.0f;

    const unsigned short* arow[4];
    const unsigned short* brow[4];
    for (int i = 0; i < 4; i++) {
        arow[i] = A + (size_t)(mBase + i * 16 + l15) * GK;
        brow[i] = W + (size_t)(nBase + i * 16 + l15) * GK;
    }

    for (int k0 = 0; k0 < GK; k0 += 32) {
        const int pi = (k0 >> 5) & 3;
        __builtin_prefetch(arow[pi] + k0 + 512, 0, 1);
        __builtin_prefetch(brow[pi] + k0 + 512, 0, 1);

        bf16x16 af[4], bfr[4];
        for (int i = 0; i < 4; i++) af[i]  = load_frag(arow[i] + k0 + kb);
        for (int i = 0; i < 4; i++) bfr[i] = load_frag(brow[i] + k0 + kb);
        for (int mi = 0; mi < 4; mi++)
            for (int ni = 0; ni < 4; ni++)
                acc[mi][ni] = wmma_bf16(af[mi], bfr[ni], acc[mi][ni]);
    }

    const int rHi = (lane >> 4) * 8;
    if (F32OUT) {
        float* base = (float*)Cout + (size_t)(mBase + rHi) * GN + nBase + l15;
        for (int mi = 0; mi < 4; mi++)
            for (int r = 0; r < 8; r++) {
                float* p = base + (mi * 16 + r) * GN;
                for (int ni = 0; ni < 4; ni++) p[ni * 16] = acc[mi][ni][r];
            }
    } else {
        unsigned short* base =
            (unsigned short*)Cout + (size_t)(mBase + rHi) * GN + nBase + l15;
        for (int mi = 0; mi < 4; mi++)
            for (int r = 0; r < 8; r++) {
                unsigned short* p = base + (mi * 16 + r) * GN;
                for (int ni = 0; ni < 4; ni++) p[ni * 16] = f2bf(acc[mi][ni][r]);
            }
    }
}

// ---------------------------------------------------------------------------
// Differential flash attention, two sequential passes (one per diff-half).
// K chunk staged by the Tensor Data Mover (wave 0 issues one descriptor).
// grid = B * H * (T/128); block = 256 threads = 8 waves, wave = 16 T-rows.
// ---------------------------------------------------------------------------
__global__ __launch_bounds__(256) void diff_flash(
    const unsigned short* __restrict__ Q, const unsigned short* __restrict__ Km,
    const unsigned short* __restrict__ Vm,
    const float* __restrict__ lq1, const float* __restrict__ lk1,
    const float* __restrict__ lq2, const float* __restrict__ lk2,
    const float* __restrict__ g, unsigned short* __restrict__ O)
{
    __shared__ unsigned short Kc[32][80];      // K chunk (one half), padded
    __shared__ unsigned short Vt[128][40];     // V chunk transposed [ch][s]
    __shared__ unsigned short Pst[8][16][40];  // per-wave P staging
    __shared__ float O1st[8][16][128];         // per-wave half-0 result

    const int TB = T_DIM / 128;
    int blk = blockIdx.x;
    const int tb = blk % TB;  blk /= TB;
    const int h  = blk % H_NUM;
    const int b  = blk / H_NUM;

    const int lane = threadIdx.x & 31;
    const int wave = threadIdx.x >> 5;
    const int l15  = lane & 15;
    const int hi   = lane >> 4;
    const int kb   = hi << 3;
    const int t0   = tb * 128 + wave * 16;
    const unsigned kcOff = (unsigned)(size_t)(AS3 void*)&Kc[0][0];

    float d1 = 0.f, d2 = 0.f;
    for (int i = 0; i < 64; i++) { d1 += lq1[i] * lk1[i]; d2 += lq2[i] * lk2[i]; }
    const float lam = __expf(d1) - __expf(d2) + LAMBDA_INIT_F;
    const float oscale = 1.0f - LAMBDA_INIT_F;

    const int sEnd = (tb + 1) * 128;

    for (int hf = 0; hf < 2; hf++) {
        const unsigned short* qrow =
            Q + (size_t)(b * T_DIM + t0 + l15) * E_DIM + (2 * h + hf) * 64;
        bf16x16 qa0 = load_frag(qrow + kb);
        bf16x16 qa1 = load_frag(qrow + 32 + kb);

        float mM[8], lL[8];
        f32x8 acc[8];
        for (int r = 0; r < 8; r++) {
            mM[r] = -1e30f; lL[r] = 0.f;
            for (int ni = 0; ni < 8; ni++) acc[ni][r] = 0.f;
        }

        for (int s0 = 0; s0 < sEnd; s0 += 32) {
            __syncthreads();   // previous chunk's LDS reads complete
            // ---- K stage: TDM descriptor (wave 0), async/sync fallback ----
#if USE_TDM
            if (wave == 0) {
                const unsigned short* src =
                    Km + (size_t)(b * T_DIM + s0) * E_DIM + (2 * h + hf) * 64;
                tdm_load_k(src, kcOff);
            }
#else
            {
                int srow = threadIdx.x >> 3;
                int col  = (threadIdx.x & 7) * 8;
                const unsigned short* src =
                    Km + (size_t)(b * T_DIM + s0 + srow) * E_DIM +
                    (2 * h + hf) * 64 + col;
                async_cp16(src, &Kc[srow][col]);
            }
#endif
            // ---- V stage, transposed into LDS: Vt[ch][s] ----
            {
                int srow = threadIdx.x & 31;
                int ch0  = (threadIdx.x >> 5) * 16;
                const unsigned short* src =
                    Vm + (size_t)(b * T_DIM + s0 + srow) * E_DIM + h * 128 + ch0;
                unsigned short tmp[16];
                *(uint4*)tmp       = *(const uint4*)src;
                *(uint4*)(tmp + 8) = *(const uint4*)(src + 8);
                for (int j = 0; j < 16; j++) Vt[ch0 + j][srow] = tmp[j];
            }
#if USE_TDM
            if (wave == 0) tdm_wait_all();
#else
            async_wait_all();
#endif
            __syncthreads();

            if (s0 <= t0 + 15) {   // wave-uniform causal skip
                // ---- S = (Q/sqrt(D)) K^T: preload all 4 B-frags, then WMMA ----
                bf16x16 kf[2][2];
                for (int sub = 0; sub < 2; sub++) {
                    const unsigned short* krow = &Kc[sub * 16 + l15][0];
                    kf[sub][0] = load_frag(krow + kb);
                    kf[sub][1] = load_frag(krow + 32 + kb);
                }
                f32x8 S[2];
                for (int sub = 0; sub < 2; sub++) {
                    f32x8 z;
                    for (int r = 0; r < 8; r++) z[r] = 0.f;
                    z = wmma_bf16(qa0, kf[sub][0], z);
                    z = wmma_bf16(qa1, kf[sub][1], z);
                    for (int r = 0; r < 8; r++) {
                        int scol = s0 + sub * 16 + l15;
                        int trow = t0 + hi * 8 + r;
                        S[sub][r] = (scol <= trow) ? z[r] * 0.125f : -1e30f;
                    }
                }
                // ---- online softmax ----
                float scl[8];
                for (int r = 0; r < 8; r++) {
                    float v = redmax16(fmaxf(S[0][r], S[1][r]));
                    float mnew = fmaxf(mM[r], v);
                    scl[r] = __expf(mM[r] - mnew);
                    mM[r] = mnew;
                    float p0 = __expf(S[0][r] - mnew);
                    float p1 = __expf(S[1][r] - mnew);
                    S[0][r] = p0; S[1][r] = p1;
                    lL[r] = lL[r] * scl[r] + redsum16(p0 + p1);
                }
                for (int ni = 0; ni < 8; ni++)
                    for (int r = 0; r < 8; r++) acc[ni][r] *= scl[r];
                // ---- P -> LDS bf16 -> A-layout fragment ----
                for (int sub = 0; sub < 2; sub++)
                    for (int r = 0; r < 8; r++)
                        Pst[wave][hi * 8 + r][sub * 16 + l15] = f2bf(S[sub][r]);
                bf16x16 pa = load_frag(&Pst[wave][l15][0] + kb);
                // ---- O += P * V ----
                for (int ni = 0; ni < 8; ni++) {
                    bf16x16 vb = load_frag(&Vt[ni * 16 + l15][0] + kb);
                    acc[ni] = wmma_bf16(pa, vb, acc[ni]);
                }
            }
        }

        // ---- per-pass finalize ----
        float inv[8];
        for (int r = 0; r < 8; r++) inv[r] = 1.0f / lL[r];
        if (hf == 0) {
            for (int ni = 0; ni < 8; ni++)
                for (int r = 0; r < 8; r++)
                    O1st[wave][hi * 8 + r][ni * 16 + l15] = acc[ni][r] * inv[r];
        } else {
            f32x8 outv[8];
            for (int ni = 0; ni < 8; ni++)
                for (int r = 0; r < 8; r++)
                    outv[ni][r] = O1st[wave][hi * 8 + r][ni * 16 + l15] -
                                  lam * acc[ni][r] * inv[r];
            float rms[8];
            for (int r = 0; r < 8; r++) {
                float s = 0.f;
                for (int ni = 0; ni < 8; ni++) s += outv[ni][r] * outv[ni][r];
                s = redsum16(s);
                rms[r] = rsqrtf(s * (1.0f / 128.0f) + 1e-5f);
            }
            unsigned short* obase =
                O + (size_t)(b * T_DIM + t0 + hi * 8) * E_DIM + h * 128 + l15;
            for (int ni = 0; ni < 8; ni++) {
                float gch = g[ni * 16 + l15];
                for (int r = 0; r < 8; r++)
                    obase[r * E_DIM + ni * 16] =
                        f2bf(outv[ni][r] * rms[r] * gch * oscale);
            }
        }
    }
}

// ---------------------------------------------------------------------------
extern "C" void kernel_launch(void* const* d_in, const int* in_sizes, int n_in,
                              void* d_out, int out_size, void* d_ws, size_t ws_size,
                              hipStream_t stream) {
    (void)n_in; (void)out_size; (void)ws_size;
    const float* X   = (const float*)d_in[0];
    const float* Wq  = (const float*)d_in[1];
    const float* Wk  = (const float*)d_in[2];
    const float* Wv  = (const float*)d_in[3];
    const float* Wo  = (const float*)d_in[4];
    const float* lq1 = (const float*)d_in[5];
    const float* lk1 = (const float*)d_in[6];
    const float* lq2 = (const float*)d_in[7];
    const float* lk2 = (const float*)d_in[8];
    const float* g   = (const float*)d_in[9];
    float* out = (float*)d_out;

    const int M = in_sizes[0] / E_DIM;    // B*T = 4096
    const int Bb = M / T_DIM;             // 2
    const int N = E_DIM, K = E_DIM;

    char* ws = (char*)d_ws;
    size_t o = 0;
    unsigned short* Xbf = (unsigned short*)(ws + o); o += (size_t)M * K * 2;
    unsigned short* Wbf = (unsigned short*)(ws + o); o += (size_t)N * K * 2;
    unsigned short* Qbf = (unsigned short*)(ws + o); o += (size_t)M * N * 2;
    unsigned short* Kbf = (unsigned short*)(ws + o); o += (size_t)M * N * 2;
    unsigned short* Vbf = (unsigned short*)(ws + o); o += (size_t)M * N * 2;
    unsigned short* Obf = (unsigned short*)(ws + o); o += (size_t)M * N * 2;

    const int nX = M * K, nW = N * K;
    dim3 cg((nX + 255) / 256), cb(256);
    dim3 wg((nW + 255) / 256);
    dim3 gg(N / 128, M / 128), gb(128);

    cast_f32_bf16<<<cg, cb, 0, stream>>>(X, Xbf, nX);

    cast_f32_bf16<<<wg, cb, 0, stream>>>(Wq, Wbf, nW);
    gemm_bf16<0><<<gg, gb, 0, stream>>>(Xbf, Wbf, (void*)Qbf);

    cast_f32_bf16<<<wg, cb, 0, stream>>>(Wk, Wbf, nW);
    gemm_bf16<0><<<gg, gb, 0, stream>>>(Xbf, Wbf, (void*)Kbf);

    cast_f32_bf16<<<wg, cb, 0, stream>>>(Wv, Wbf, nW);
    gemm_bf16<0><<<gg, gb, 0, stream>>>(Xbf, Wbf, (void*)Vbf);

    diff_flash<<<Bb * H_NUM * (T_DIM / 128), 256, 0, stream>>>(
        Qbf, Kbf, Vbf, lq1, lk1, lq2, lk2, g, Obf);

    cast_f32_bf16<<<wg, cb, 0, stream>>>(Wo, Wbf, nW);
    gemm_bf16<1><<<gg, gb, 0, stream>>>(Obf, Wbf, (void*)out);
}
